// MoELayer_52432960749839
// MI455X (gfx1250) — matrix operand order repaired and checked by reading, not compiled, not run on hardware
//
#include <hip/hip_runtime.h>
#include <hip/hip_bf16.h>

// Problem constants (B=2, T=2048 -> N=4096 passed at runtime)
#define HDIM  512
#define EDIM  8
#define ITER  1024
#define NSEG  9          // 8 routed experts + 1 shared "expert"

typedef __attribute__((ext_vector_type(16))) __bf16 v16bf;
typedef __attribute__((ext_vector_type(8)))  float  v8f;

union Frag16 {
    v16bf v;
    uint4 q[2];
};

// ---------------------------------------------------------------------------
// Router: scores = softmax(x @ router_w), top-2, renormalized weights.
// ---------------------------------------------------------------------------
__global__ void router_k(const float* __restrict__ x, const float* __restrict__ rw,
                         int* __restrict__ cnt, int* __restrict__ tokE,
                         float* __restrict__ tokW, int N) {
    __shared__ float rws[HDIM * EDIM];
    for (int i = threadIdx.x; i < HDIM * EDIM; i += blockDim.x) rws[i] = rw[i];
    __syncthreads();

    int n = blockIdx.x * blockDim.x + threadIdx.x;
    if (n >= N) return;

    float s[EDIM] = {0.f, 0.f, 0.f, 0.f, 0.f, 0.f, 0.f, 0.f};
    const float* xr = x + (size_t)n * HDIM;
    for (int h = 0; h < HDIM; ++h) {
        float xv = xr[h];
#pragma unroll
        for (int e = 0; e < EDIM; ++e) s[e] = fmaf(xv, rws[h * EDIM + e], s[e]);
    }
    float mx = s[0];
#pragma unroll
    for (int e = 1; e < EDIM; ++e) mx = fmaxf(mx, s[e]);
    float sum = 0.f;
#pragma unroll
    for (int e = 0; e < EDIM; ++e) { s[e] = __expf(s[e] - mx); sum += s[e]; }
    float inv = 1.0f / sum;
#pragma unroll
    for (int e = 0; e < EDIM; ++e) s[e] *= inv;
    int e0 = 0; float p0 = s[0];
#pragma unroll
    for (int e = 1; e < EDIM; ++e) if (s[e] > p0) { p0 = s[e]; e0 = e; }
    int e1 = (e0 == 0) ? 1 : 0; float p1 = s[e1];
#pragma unroll
    for (int e = 0; e < EDIM; ++e)
        if (e != e0 && s[e] > p1) { p1 = s[e]; e1 = e; }

    float rn = 1.0f / (p0 + p1);
    tokE[2 * n] = e0;      tokE[2 * n + 1] = e1;
    tokW[2 * n] = p0 * rn; tokW[2 * n + 1] = p1 * rn;
    atomicAdd(&cnt[e0], 1);
    atomicAdd(&cnt[e1], 1);
}

__global__ void scan_k(const int* __restrict__ cnt, int* __restrict__ base,
                       int* __restrict__ cnt2, int N) {
    if (threadIdx.x == 0) {
        int acc = 0;
#pragma unroll
        for (int e = 0; e < EDIM; ++e) { base[e] = acc; acc += cnt[e]; }
        base[EDIM] = acc;           // start of shared segment (== 2N)
        base[EDIM + 1] = acc + N;   // end of shared segment
#pragma unroll
        for (int e = 0; e < 16; ++e) cnt2[e] = 0;
    }
}

__global__ void scatter_k(const int* __restrict__ tokE, const float* __restrict__ tokW,
                          const int* __restrict__ base, int* __restrict__ cnt2,
                          int* __restrict__ perm, float* __restrict__ ewt, int N) {
    int n = blockIdx.x * blockDim.x + threadIdx.x;
    if (n >= N) return;
#pragma unroll
    for (int k = 0; k < 2; ++k) {
        int e = tokE[2 * n + k];
        int pos = base[e] + atomicAdd(&cnt2[e], 1);
        perm[pos] = n;
        ewt[pos] = tokW[2 * n + k];
    }
    perm[base[EDIM] + n] = n;      // shared expert entry
    ewt[base[EDIM] + n] = 1.0f;
}

__global__ void cast_x_k(const float* __restrict__ x, __bf16* __restrict__ xh, int total) {
    int i = blockIdx.x * blockDim.x + threadIdx.x;
    if (i < total) xh[i] = (__bf16)x[i];
}

// Cast + transpose weights to bf16, [seg][n][k]; segment 8 = shared expert.
__global__ void cast_w_k(const float* __restrict__ Wg, const float* __restrict__ Wu,
                         const float* __restrict__ Wd, const float* __restrict__ Sg,
                         const float* __restrict__ Su, const float* __restrict__ Sd,
                         __bf16* __restrict__ WgT, __bf16* __restrict__ WuT,
                         __bf16* __restrict__ WdT) {
    size_t idx = (size_t)blockIdx.x * blockDim.x + threadIdx.x;
    const size_t tot = (size_t)NSEG * ITER * HDIM;
    if (idx >= tot) return;
    int h = (int)(idx % HDIM);
    int i = (int)((idx / HDIM) % ITER);
    int e = (int)(idx / ((size_t)HDIM * ITER));
    float g, u, d;
    if (e < EDIM) {
        g = Wg[((size_t)e * HDIM + h) * ITER + i];
        u = Wu[((size_t)e * HDIM + h) * ITER + i];
        d = Wd[((size_t)e * ITER + i) * HDIM + h];
    } else {
        g = Sg[(size_t)h * ITER + i];
        u = Su[(size_t)h * ITER + i];
        d = Sd[(size_t)i * HDIM + h];
    }
    WgT[idx] = (__bf16)g;                               // [e][i][h]: n=i, k=h
    WuT[idx] = (__bf16)u;
    WdT[((size_t)e * HDIM + h) * ITER + i] = (__bf16)d; // [e][h][i]: n=h, k=i
}

// ---------------------------------------------------------------------------
// GEMM1: h = silu(x@Wg)*(x@Wu). Block = 8 waves; block tile = 32 tokens x
// 128 inter-cols; each wave: 32 tokens x 16 cols (2 A-frags from LDS reuse
// one Bg/Bu fragment pair -> 4 WMMA per 4 ds + 4 vmem loads).
// ---------------------------------------------------------------------------
__global__ __launch_bounds__(256) void gemm1_k(
    const __bf16* __restrict__ xh, const __bf16* __restrict__ WgT,
    const __bf16* __restrict__ WuT, const int* __restrict__ base,
    const int* __restrict__ perm, __bf16* __restrict__ hbuf) {
    const int e = blockIdx.y;
    const int seg = base[e];
    const int count = base[e + 1] - seg;
    const int m0 = blockIdx.x * 32;
    if (m0 >= count) return;

    __shared__ __bf16 As[32][520];   // pitch 520 (1040 B): conflict-free b128 reads
    {   // gather 32 permuted token rows (each thread: one 64-col chunk)
        int t = threadIdx.x;
        int r = t >> 3;
        int c0 = (t & 7) * 64;
        int row = m0 + r;
        uint4* dst = reinterpret_cast<uint4*>(&As[r][c0]);
        if (row < count) {
            int tok = perm[seg + row];
            const uint4* src = reinterpret_cast<const uint4*>(xh + (size_t)tok * HDIM + c0);
#pragma unroll
            for (int i = 0; i < 8; ++i) dst[i] = src[i];
        } else {
            uint4 z = make_uint4(0u, 0u, 0u, 0u);
#pragma unroll
            for (int i = 0; i < 8; ++i) dst[i] = z;
        }
    }
    __syncthreads();

    const int lane = threadIdx.x & 31;
    const int w = threadIdx.x >> 5;               // 8 waves -> 8 col groups
    const int ncol0 = blockIdx.z * 128 + w * 16;
    const int lr = lane & 15;
    const int hi = lane >> 4;
    const int kbA = hi * 8;                       // A: hi half holds K 8..15 / 24..31
    const int khB = hi * 8;                       // B: hi half holds K 8..15 / 24..31

    v8f accg0 = {}, accg1 = {}, accu0 = {}, accu1 = {};
    const __bf16* bgrow = WgT + ((size_t)e * ITER + ncol0 + lr) * HDIM + khB;
    const __bf16* burow = WuT + ((size_t)e * ITER + ncol0 + lr) * HDIM + khB;
    const __bf16* a0lo = &As[lr][kbA];            // token rows 0..15
    const __bf16* a0hi = &As[lr][kbA + 16];
    const __bf16* a1lo = &As[16 + lr][kbA];       // token rows 16..31
    const __bf16* a1hi = &As[16 + lr][kbA + 16];

#pragma unroll 2
    for (int k0 = 0; k0 < HDIM; k0 += 32) {
        Frag16 a0, a1, bg, bu;
        a0.q[0] = *reinterpret_cast<const uint4*>(a0lo + k0);
        a0.q[1] = *reinterpret_cast<const uint4*>(a0hi + k0);
        a1.q[0] = *reinterpret_cast<const uint4*>(a1lo + k0);
        a1.q[1] = *reinterpret_cast<const uint4*>(a1hi + k0);
        bg.q[0] = *reinterpret_cast<const uint4*>(bgrow + k0);
        bg.q[1] = *reinterpret_cast<const uint4*>(bgrow + k0 + 16);
        bu.q[0] = *reinterpret_cast<const uint4*>(burow + k0);
        bu.q[1] = *reinterpret_cast<const uint4*>(burow + k0 + 16);
        accg0 = __builtin_amdgcn_wmma_f32_16x16x32_bf16(false, a0.v, false, bg.v,
                                                        (short)0, accg0, false, false);
        accg1 = __builtin_amdgcn_wmma_f32_16x16x32_bf16(false, a1.v, false, bg.v,
                                                        (short)0, accg1, false, false);
        accu0 = __builtin_amdgcn_wmma_f32_16x16x32_bf16(false, a0.v, false, bu.v,
                                                        (short)0, accu0, false, false);
        accu1 = __builtin_amdgcn_wmma_f32_16x16x32_bf16(false, a1.v, false, bu.v,
                                                        (short)0, accu1, false, false);
    }

#pragma unroll
    for (int rg = 0; rg < 2; ++rg) {
        v8f ag = rg ? accg1 : accg0;
        v8f au = rg ? accu1 : accu0;
#pragma unroll
        for (int r = 0; r < 8; ++r) {
            int m = rg * 16 + hi * 8 + r;         // C/D: VGPR r -> M = r + 8*hi
            int row = m0 + m;
            if (row < count) {
                float g = ag[r], u = au[r];
                // silu(g)*u with fast reciprocal (v_rcp_f32) instead of IEEE div
                float hv = g * u * __builtin_amdgcn_rcpf(1.0f + __expf(-g));
                hbuf[(size_t)(seg + row) * ITER + ncol0 + lr] = (__bf16)hv;
            }
        }
    }
}

// ---------------------------------------------------------------------------
// GEMM2: out[tok, h] += ewt * (h @ Wd). Block = 8 waves x (32 rows x 32
// cols) = 32 rows x 256 H-cols per block; 4 WMMA per 8 b128 loads.
// ---------------------------------------------------------------------------
__global__ __launch_bounds__(256) void gemm2_k(
    const __bf16* __restrict__ hbuf, const __bf16* __restrict__ WdT,
    const int* __restrict__ base, const int* __restrict__ perm,
    const float* __restrict__ ewt, float* __restrict__ out) {
    const int e = blockIdx.y;
    const int seg = base[e];
    const int count = base[e + 1] - seg;
    const int m0 = blockIdx.x * 32;
    if (m0 >= count) return;

    const int lane = threadIdx.x & 31;
    const int w = threadIdx.x >> 5;
    const int cb = blockIdx.z * 256 + w * 32;     // this wave's 32 H-columns
    const int lr = lane & 15;
    const int hi = lane >> 4;
    const int kbA = hi * 8;
    const int khB = hi * 8;

    int r0 = m0 + lr;       if (r0 >= count) r0 = count - 1;   // clamped; results discarded
    int r1 = m0 + 16 + lr;  if (r1 >= count) r1 = count - 1;
    const __bf16* arow0 = hbuf + (size_t)(seg + r0) * ITER + kbA;
    const __bf16* arow1 = hbuf + (size_t)(seg + r1) * ITER + kbA;
    const __bf16* brow0 = WdT + ((size_t)e * HDIM + cb + lr) * ITER + khB;
    const __bf16* brow1 = WdT + ((size_t)e * HDIM + cb + 16 + lr) * ITER + khB;

    v8f acc00 = {}, acc01 = {}, acc10 = {}, acc11 = {};
#pragma unroll 2
    for (int k0 = 0; k0 < ITER; k0 += 32) {
        Frag16 a0, a1, b0, b1;
        a0.q[0] = *reinterpret_cast<const uint4*>(arow0 + k0);
        a0.q[1] = *reinterpret_cast<const uint4*>(arow0 + k0 + 16);
        a1.q[0] = *reinterpret_cast<const uint4*>(arow1 + k0);
        a1.q[1] = *reinterpret_cast<const uint4*>(arow1 + k0 + 16);
        b0.q[0] = *reinterpret_cast<const uint4*>(brow0 + k0);
        b0.q[1] = *reinterpret_cast<const uint4*>(brow0 + k0 + 16);
        b1.q[0] = *reinterpret_cast<const uint4*>(brow1 + k0);
        b1.q[1] = *reinterpret_cast<const uint4*>(brow1 + k0 + 16);
        acc00 = __builtin_amdgcn_wmma_f32_16x16x32_bf16(false, a0.v, false, b0.v,
                                                        (short)0, acc00, false, false);
        acc01 = __builtin_amdgcn_wmma_f32_16x16x32_bf16(false, a0.v, false, b1.v,
                                                        (short)0, acc01, false, false);
        acc10 = __builtin_amdgcn_wmma_f32_16x16x32_bf16(false, a1.v, false, b0.v,
                                                        (short)0, acc10, false, false);
        acc11 = __builtin_amdgcn_wmma_f32_16x16x32_bf16(false, a1.v, false, b1.v,
                                                        (short)0, acc11, false, false);
    }

#pragma unroll
    for (int rg = 0; rg < 2; ++rg) {
#pragma unroll
        for (int cg = 0; cg < 2; ++cg) {
            v8f acc = rg ? (cg ? acc11 : acc10) : (cg ? acc01 : acc00);
#pragma unroll
            for (int r = 0; r < 8; ++r) {
                int m = rg * 16 + hi * 8 + r;
                int row = m0 + m;
                if (row < count) {
                    int ent = seg + row;
                    int tok = perm[ent];
                    float wt = ewt[ent];
                    atomicAdd(&out[(size_t)tok * HDIM + cb + cg * 16 + lr], wt * acc[r]);
                }
            }
        }
    }
}

// ---------------------------------------------------------------------------
extern "C" void kernel_launch(void* const* d_in, const int* in_sizes, int n_in,
                              void* d_out, int out_size, void* d_ws, size_t ws_size,
                              hipStream_t stream) {
    (void)n_in; (void)ws_size;
    const float* x  = (const float*)d_in[0];
    const float* rw = (const float*)d_in[1];
    const float* Wg = (const float*)d_in[2];
    const float* Wu = (const float*)d_in[3];
    const float* Wd = (const float*)d_in[4];
    const float* Sg = (const float*)d_in[5];
    const float* Su = (const float*)d_in[6];
    const float* Sd = (const float*)d_in[7];
    float* out = (float*)d_out;

    const int N = in_sizes[0] / HDIM;             // 4096 tokens

    char* ws = (char*)d_ws;
    size_t off = 0;
    auto alloc = [&](size_t bytes) {
        char* p = ws + off;
        off = (off + bytes + 255) & ~(size_t)255;
        return p;
    };
    int*    cnt  = (int*)alloc(64);
    int*    basep= (int*)alloc(64);
    int*    cnt2 = (int*)alloc(64);
    int*    tokE = (int*)alloc(sizeof(int) * 2 * (size_t)N);
    float*  tokW = (float*)alloc(sizeof(float) * 2 * (size_t)N);
    int*    perm = (int*)alloc(sizeof(int) * 3 * (size_t)N);
    float*  ewtp = (float*)alloc(sizeof(float) * 3 * (size_t)N);
    __bf16* xh   = (__bf16*)alloc(sizeof(__bf16) * (size_t)N * HDIM);
    const size_t wtot = (size_t)NSEG * ITER * HDIM;
    __bf16* WgT  = (__bf16*)alloc(sizeof(__bf16) * wtot);
    __bf16* WuT  = (__bf16*)alloc(sizeof(__bf16) * wtot);
    __bf16* WdT  = (__bf16*)alloc(sizeof(__bf16) * wtot);
    __bf16* hbuf = (__bf16*)alloc(sizeof(__bf16) * 3 * (size_t)N * ITER);

    hipMemsetAsync(cnt, 0, 64, stream);
    hipMemsetAsync(out, 0, sizeof(float) * (size_t)out_size, stream);

    router_k<<<(N + 255) / 256, 256, 0, stream>>>(x, rw, cnt, tokE, tokW, N);
    scan_k<<<1, 32, 0, stream>>>(cnt, basep, cnt2, N);
    scatter_k<<<(N + 255) / 256, 256, 0, stream>>>(tokE, tokW, basep, cnt2, perm, ewtp, N);

    const size_t xtot = (size_t)N * HDIM;
    cast_x_k<<<(int)((xtot + 255) / 256), 256, 0, stream>>>(x, xh, (int)xtot);
    cast_w_k<<<(int)((wtot + 255) / 256), 256, 0, stream>>>(Wg, Wu, Wd, Sg, Su, Sd,
                                                            WgT, WuT, WdT);

    dim3 g1((N + 31) / 32, NSEG, ITER / 128);
    gemm1_k<<<g1, 256, 0, stream>>>(xh, WgT, WuT, basep, perm, hbuf);

    dim3 g2((N + 31) / 32, NSEG, HDIM / 256);
    gemm2_k<<<g2, 256, 0, stream>>>(hbuf, WdT, basep, perm, ewtp, out);
}
